// OutlookAttention_26895085207896
// MI455X (gfx1250) — compile-verified, matrix operand order checked by reading, not run on hardware
//
#include <hip/hip_runtime.h>
#include <hip/hip_bf16.h>

typedef __attribute__((ext_vector_type(16))) _Float16 v16h;
typedef __attribute__((ext_vector_type(8)))  float    v8f;
typedef __attribute__((ext_vector_type(4)))  float    v4f;

#define BATCH   8
#define HH      56
#define WW      56
#define CC      384
#define NPIX    (BATCH*HH*WW)        // 25088 rows, 1568 tiles of 16
#define NA      81                   // K^4 attention logits
#define NCOLS   (CC + NA)            // 465
#define NTILES  30                   // ceil(465/16) -> 480 padded
#define KTILES  (CC/32)              // 12
#define NT_PER_WAVE 5                // 6 wave-groups * 5 tiles = 30

// ---------------- workspace layout ----------------
// Wh   : f16 packed weights [30][12][32][16]          = 368640 B
// bias : f32 [480]                                    = 1920 B
// v    : f32 [25088][384]                             = 38535168 B
// attn : f32 [25088][81]                              = 8128512 B
#define WS_WH    0
#define WS_BIAS  368640
#define WS_V     370560
#define WS_ATTN  (370560 + 38535168)

// ============ kernel 0: pack weights into WMMA B-fragment order ============
__global__ void oa_pack_kernel(const float* __restrict__ Wv,
                               const float* __restrict__ bv,
                               const float* __restrict__ Wa,
                               const float* __restrict__ ba,
                               _Float16* __restrict__ Wh,
                               float* __restrict__ bias) {
    int idx = blockIdx.x * 256 + threadIdx.x;
    if (idx < 480) {
        bias[idx] = (idx < CC) ? bv[idx] : ((idx < NCOLS) ? ba[idx - CC] : 0.f);
    }
    if (idx >= NTILES * KTILES * 32 * 16) return;
    int e    = idx & 15;
    int lane = (idx >> 4) & 31;
    int kt   = (idx >> 9) % KTILES;
    int nt   = (idx >> 9) / KTILES;
    int h = lane >> 4, n = lane & 15;
    // per-lane K set: {8h..8h+7} U {16+8h..16+8h+7} within the 32-wide k-tile
    int k = kt * 32 + ((e < 8) ? (8 * h + e) : (16 + 8 * h + (e - 8)));
    int col = nt * 16 + n;
    float w;
    if (col < CC)          w = Wv[k * CC + col];
    else if (col < NCOLS)  w = Wa[k * NA + (col - CC)];
    else                   w = 0.f;
    Wh[idx] = (_Float16)w;
}

// ============ kernel 1: fused v/attn projection GEMM via WMMA ============
__global__ __launch_bounds__(32)
void oa_gemm_kernel(const float* __restrict__ img,
                    const _Float16* __restrict__ Wh,
                    const float* __restrict__ bias,
                    float* __restrict__ v_out,
                    float* __restrict__ attn_out) {
    const int mtile = blockIdx.x;        // 0..1567
    const int ngrp  = blockIdx.y;        // 0..5
    const int lane  = threadIdx.x;       // wave32
    const int h = lane >> 4;             // lane half selects K sub-range
    const int r = lane & 15;             // A row / D column index

    const float* arow = img + (size_t)(mtile * 16 + r) * CC;

    v8f acc[NT_PER_WAVE];
#pragma unroll
    for (int t = 0; t < NT_PER_WAVE; ++t)
#pragma unroll
        for (int i = 0; i < 8; ++i) acc[t][i] = 0.f;

    for (int kt = 0; kt < KTILES; ++kt) {
        const int k0 = kt * 32 + 8 * h;
        v4f a0 = *(const v4f*)(arow + k0);
        v4f a1 = *(const v4f*)(arow + k0 + 4);
        v4f a2 = *(const v4f*)(arow + k0 + 16);
        v4f a3 = *(const v4f*)(arow + k0 + 20);
        v16h a;
        a[0]=(_Float16)a0.x;  a[1]=(_Float16)a0.y;  a[2]=(_Float16)a0.z;  a[3]=(_Float16)a0.w;
        a[4]=(_Float16)a1.x;  a[5]=(_Float16)a1.y;  a[6]=(_Float16)a1.z;  a[7]=(_Float16)a1.w;
        a[8]=(_Float16)a2.x;  a[9]=(_Float16)a2.y;  a[10]=(_Float16)a2.z; a[11]=(_Float16)a2.w;
        a[12]=(_Float16)a3.x; a[13]=(_Float16)a3.y; a[14]=(_Float16)a3.z; a[15]=(_Float16)a3.w;
#pragma unroll
        for (int t = 0; t < NT_PER_WAVE; ++t) {
            const int nt = ngrp * NT_PER_WAVE + t;
            v16h b = *(const v16h*)(Wh + (((size_t)(nt * KTILES + kt) * 32 + lane) << 4));
            acc[t] = __builtin_amdgcn_wmma_f32_16x16x32_f16(
                false, a, false, b, (short)0, acc[t], false, false);
        }
    }

    // D layout: VGPR i, lane l -> M = i + 8*(l/16), N = l%16
#pragma unroll
    for (int t = 0; t < NT_PER_WAVE; ++t) {
        const int col = (ngrp * NT_PER_WAVE + t) * 16 + r;
        if (col >= NCOLS) continue;
        const float bb = bias[col];
#pragma unroll
        for (int i = 0; i < 8; ++i) {
            const size_t mrow = (size_t)mtile * 16 + i + 8 * h;
            const float val = acc[t][i] + bb;
            if (col < CC) v_out[mrow * CC + col] = val;
            else          attn_out[mrow * NA + (col - CC)] = val;
        }
    }
}

// ============ kernel 2: softmax over q (rows of 9) ============
__global__ void oa_softmax_kernel(float* __restrict__ attn) {
    int idx = blockIdx.x * 256 + threadIdx.x;     // pixel*9 + p
    if (idx >= NPIX * 9) return;
    float* a = attn + (size_t)idx * 9;            // [pix][p][q] : contiguous 9
    float m = a[0];
#pragma unroll
    for (int q = 1; q < 9; ++q) m = fmaxf(m, a[q]);
    float e[9], s = 0.f;
#pragma unroll
    for (int q = 0; q < 9; ++q) { e[q] = __expf(a[q] - m); s += e[q]; }
    float inv = 1.f / s;
#pragma unroll
    for (int q = 0; q < 9; ++q) a[q] = e[q] * inv;
}

// ============ kernel 3: weighted aggregation + fold (gather form) ============
// out[y,x,c] = sum_{i,j in 3x3} sum_{q} attn[h=y+1-i, w=x+1-j][(i,j)][q] * v[h-1+qi, w-1+qj, c]
#define TS 8      // spatial tile
#define CH 96     // channels per block (4 chunks)
__global__ __launch_bounds__(256)
void oa_agg_kernel(const float* __restrict__ v,
                   const float* __restrict__ attn,
                   float* __restrict__ out) {
    __shared__ float v_s[12 * 12 * CH];           // 55296 B halo tile
    const int tile  = blockIdx.x;                 // 49 tiles
    const int ty = tile / (WW / TS), tx = tile % (WW / TS);
    const int c0 = blockIdx.y * CH;               // channel chunk
    const int b  = blockIdx.z;
    const int t  = threadIdx.x;

    const int y0 = ty * TS - 2, x0 = tx * TS - 2; // halo origin
    for (int i = t; i < 12 * 12 * CH; i += 256) {
        int c = i % CH, rest = i / CH;
        int lx = rest % 12, ly = rest / 12;
        int gy = y0 + ly, gx = x0 + lx;
        float val = 0.f;
        if (gy >= 0 && gy < HH && gx >= 0 && gx < WW)
            val = v[((size_t)(b * HH + gy) * WW + gx) * CC + c0 + c];
        v_s[i] = val;
    }
    __syncthreads();

    const int pix = t >> 2, ci = t & 3;           // 64 pixels x 4 channel lanes
    const int ly = pix >> 3, lx = pix & 7;
    const int y = ty * TS + ly, x = tx * TS + lx;

    float acc[CH / 4];
#pragma unroll
    for (int cc = 0; cc < CH / 4; ++cc) acc[cc] = 0.f;

    for (int i = 0; i < 3; ++i)
        for (int j = 0; j < 3; ++j) {
            const int hc = y + 1 - i, wc = x + 1 - j;    // window center
            if (hc < 0 || hc >= HH || wc < 0 || wc >= WW) continue;
            const float* ap = attn + ((size_t)(b * HH + hc) * WW + wc) * NA + (i * 3 + j) * 9;
#pragma unroll
            for (int q = 0; q < 9; ++q) {
                const float wgt = ap[q];
                const int vy = ly - i + (q / 3) + 2;     // 0..11
                const int vx = lx - j + (q % 3) + 2;
                const float* vp = v_s + (vy * 12 + vx) * CH + ci;
#pragma unroll
                for (int cc = 0; cc < CH / 4; ++cc)
                    acc[cc] += wgt * vp[cc * 4];
            }
        }

    float* op = out + ((size_t)(b * HH + y) * WW + x) * CC + c0 + ci;
#pragma unroll
    for (int cc = 0; cc < CH / 4; ++cc) op[cc * 4] = acc[cc];
}

extern "C" void kernel_launch(void* const* d_in, const int* in_sizes, int n_in,
                              void* d_out, int out_size, void* d_ws, size_t ws_size,
                              hipStream_t stream) {
    const float* image = (const float*)d_in[0];
    const float* Wv    = (const float*)d_in[1];
    const float* bv    = (const float*)d_in[2];
    const float* Wa    = (const float*)d_in[3];
    const float* ba    = (const float*)d_in[4];
    float* out = (float*)d_out;

    char* ws = (char*)d_ws;
    _Float16* Wh   = (_Float16*)(ws + WS_WH);
    float*    bias = (float*)(ws + WS_BIAS);
    float*    vbuf = (float*)(ws + WS_V);
    float*    attn = (float*)(ws + WS_ATTN);

    // 0) pack weights into B-fragment order (+ bias concat)
    {
        int n = NTILES * KTILES * 32 * 16;
        oa_pack_kernel<<<(n + 255) / 256, 256, 0, stream>>>(Wv, bv, Wa, ba, Wh, bias);
    }
    // 1) fused projection GEMM -> v, attn logits
    {
        dim3 grid(NPIX / 16, 6);
        oa_gemm_kernel<<<grid, 32, 0, stream>>>(image, Wh, bias, vbuf, attn);
    }
    // 2) softmax over q
    {
        int n = NPIX * 9;
        oa_softmax_kernel<<<(n + 255) / 256, 256, 0, stream>>>(attn);
    }
    // 3) aggregate + fold
    {
        dim3 grid((HH / TS) * (WW / TS), CC / CH, BATCH);
        oa_agg_kernel<<<grid, 256, 0, stream>>>(vbuf, attn, out);
    }
}
